// SpectralRWKVLayer_8675833938455
// MI455X (gfx1250) — compile-verified, matrix operand order checked by reading, not run on hardware
//
#include <hip/hip_runtime.h>
#include <hip/hip_bf16.h>

// ---------------- shapes ----------------
#define BZ   4
#define TT   512
#define FF   128
#define HH   64
#define DD   8192          // F*H
#define SS   64            // d_state
#define M1   2048          // B*T
#define M2   262144        // B*T*F
#define H4   256           // H*FFN_MULT

typedef __attribute__((ext_vector_type(16))) _Float16 v16h;
typedef __attribute__((ext_vector_type(8)))  _Float16 v8h;
typedef __attribute__((ext_vector_type(8)))  float    v8f;

#define WMMA_F16(A,B,C) __builtin_amdgcn_wmma_f32_16x16x32_f16(false,(A),false,(B),(short)0,(C),false,false)

__device__ __forceinline__ v16h load_frag(const _Float16* p) {
    // per-lane A/B fragment: 8 halves at K=kb, 8 halves at K=kb+16 (ISA 16-bit 16x32 layout)
    v8h c0 = *reinterpret_cast<const v8h*>(p);
    v8h c1 = *reinterpret_cast<const v8h*>(p + 16);
    v16h r;
#pragma unroll
    for (int i = 0; i < 8; ++i) { r[i] = c0[i]; r[8 + i] = c1[i]; }
    return r;
}
__device__ __forceinline__ v16h zero_frag() {
    v16h r;
#pragma unroll
    for (int i = 0; i < 16; ++i) r[i] = (_Float16)0.f;
    return r;
}
__device__ __forceinline__ v8f zero_acc() {
    v8f a;
#pragma unroll
    for (int i = 0; i < 8; ++i) a[i] = 0.f;
    return a;
}
__device__ __forceinline__ float gelu_tanh(float x) {
    float x3 = x * x * x;
    return 0.5f * x * (1.f + tanhf(0.7978845608028654f * (x + 0.044715f * x3)));
}
__device__ __forceinline__ float sigmoidf_(float x) { return 1.f / (1.f + expf(-x)); }

// ------------------------------------------------------------------
// K0: weight transpose + f32 -> f16 conversion (run once per launch)
// ------------------------------------------------------------------
__global__ void prep_weights(const float* __restrict__ Wk, const float* __restrict__ Wv,
                             const float* __restrict__ Wr, const float* __restrict__ Wo,
                             const float* __restrict__ fW, const float* __restrict__ W1,
                             const float* __restrict__ W2,
                             _Float16* __restrict__ WkT, _Float16* __restrict__ WvT,
                             _Float16* __restrict__ WrT, _Float16* __restrict__ WoT,
                             _Float16* __restrict__ fWh, _Float16* __restrict__ W1T,
                             _Float16* __restrict__ W2T)
{
    int i = blockIdx.x * 256 + threadIdx.x;
    const int NP = SS * DD;                       // 524288
    if (i < NP) { int n = i >> 13, d = i & 8191; WkT[i] = (_Float16)Wk[d * SS + n]; return; }
    i -= NP;
    if (i < NP) { int n = i >> 13, d = i & 8191; WvT[i] = (_Float16)Wv[d * SS + n]; return; }
    i -= NP;
    if (i < NP) { int n = i >> 13, d = i & 8191; WrT[i] = (_Float16)Wr[d * SS + n]; return; }
    i -= NP;
    if (i < NP) { int n = i >> 6, s = i & 63; WoT[i] = (_Float16)Wo[s * DD + n]; return; }
    i -= NP;
    if (i < 3 * HH * HH) {                        // freq_W (HO,HI,3) -> [k][ho][hi]
        int k = i / 4096, r = i % 4096, ho = r >> 6, hi = r & 63;
        fWh[i] = (_Float16)fW[ho * (HH * 3) + hi * 3 + k]; return;
    }
    i -= 3 * HH * HH;
    if (i < H4 * HH) { int n = i >> 6, kk = i & 63; W1T[i] = (_Float16)W1[kk * H4 + n]; return; }
    i -= H4 * HH;
    if (i < HH * H4) { int n = i >> 8, kk = i & 255; W2T[i] = (_Float16)W2[kk * HH + n]; return; }
}

// ------------------------------------------------------------------
// LayerNorm over d=8192 -> f16 (one workgroup per row)
// ------------------------------------------------------------------
__global__ void ln_to_f16(const float* __restrict__ x, const float* __restrict__ g,
                          const float* __restrict__ b, _Float16* __restrict__ out)
{
    __shared__ float s1[256], s2[256];
    int tid = threadIdx.x;
    size_t row = blockIdx.x;
    const float* xr = x + row * DD;
    float ps = 0.f, pq = 0.f;
    for (int d = tid; d < DD; d += 256) { float v = xr[d]; ps += v; pq += v * v; }
    s1[tid] = ps; s2[tid] = pq; __syncthreads();
    for (int off = 128; off > 0; off >>= 1) {
        if (tid < off) { s1[tid] += s1[tid + off]; s2[tid] += s2[tid + off]; }
        __syncthreads();
    }
    float mu = s1[0] * (1.f / DD);
    float rstd = rsqrtf(s2[0] * (1.f / DD) - mu * mu + 1e-5f);
    _Float16* o = out + row * DD;
    for (int d = tid; d < DD; d += 256) o[d] = (_Float16)((xr[d] - mu) * rstd * g[d] + b[d]);
}

// per-row mean / rstd only (for fused time-conv LN)
__global__ void row_stats(const float* __restrict__ x, float* __restrict__ st)
{
    __shared__ float s1[256], s2[256];
    int tid = threadIdx.x;
    size_t row = blockIdx.x;
    const float* xr = x + row * DD;
    float ps = 0.f, pq = 0.f;
    for (int d = tid; d < DD; d += 256) { float v = xr[d]; ps += v; pq += v * v; }
    s1[tid] = ps; s2[tid] = pq; __syncthreads();
    for (int off = 128; off > 0; off >>= 1) {
        if (tid < off) { s1[tid] += s1[tid + off]; s2[tid] += s2[tid + off]; }
        __syncthreads();
    }
    if (tid == 0) {
        float mu = s1[0] * (1.f / DD);
        st[row * 2]     = mu;
        st[row * 2 + 1] = rsqrtf(s2[0] * (1.f / DD) - mu * mu + 1e-5f);
    }
}

// ------------------------------------------------------------------
// K2: k/v/r projections  (2048,8192) x (8192,64) via WMMA, 12 waves/WG
// ------------------------------------------------------------------
__global__ void proj_kvr(const _Float16* __restrict__ xn,
                         const _Float16* __restrict__ WkT, const _Float16* __restrict__ WvT,
                         const _Float16* __restrict__ WrT,
                         float* __restrict__ ko, float* __restrict__ vo, float* __restrict__ ro)
{
    int lane = threadIdx.x & 31;
    int wave = threadIdx.x >> 5;      // 0..11
    int mat  = wave >> 2;             // 0:k 1:v 2:r
    int nt   = wave & 3;              // N-tile
    const _Float16* BT = (mat == 0) ? WkT : ((mat == 1) ? WvT : WrT);
    float* out = (mat == 0) ? ko : ((mat == 1) ? vo : ro);
    int row0 = blockIdx.x * 16;
    int m  = lane & 15;
    int kb = (lane >> 4) * 8;
    const _Float16* Arow = xn + (size_t)(row0 + m) * DD + kb;
    const _Float16* Brow = BT + (size_t)(nt * 16 + m) * DD + kb;
    v8f acc = zero_acc();
    for (int k0 = 0; k0 < DD; k0 += 32) {
        __builtin_prefetch(Arow + k0 + 512, 0, 0);   // global_prefetch_b8
        v16h a = load_frag(Arow + k0);
        v16h b = load_frag(Brow + k0);
        acc = WMMA_F16(a, b, acc);
    }
    int ncol = nt * 16 + m;
#pragma unroll
    for (int v = 0; v < 8; ++v) {
        int mm = v + 8 * (lane >> 4);
        out[(size_t)(row0 + mm) * SS + ncol] = acc[v];
    }
}

// ------------------------------------------------------------------
// K3: RWKV scan over T, one thread per (b, s) channel
// ------------------------------------------------------------------
__global__ void rwkv_scan(const float* __restrict__ kk, const float* __restrict__ vv,
                          const float* __restrict__ rr, const float* __restrict__ wdecay,
                          _Float16* __restrict__ rs)
{
    int s = threadIdx.x;              // 0..63
    int b = blockIdx.x;               // 0..3
    float decay = expf(-expf(wdecay[s]));
    float st = 0.f;
    size_t base = (size_t)b * TT * SS + s;
    for (int t = 0; t < TT; ++t) {
        size_t idx = base + (size_t)t * SS;
        st = decay * st + kk[idx] * vv[idx];
        rs[idx] = (_Float16)(sigmoidf_(rr[idx]) * st);
    }
}

// ------------------------------------------------------------------
// K4: attn = rs(2048,64) @ Wo(64,8192); d_out = x + attn
// ------------------------------------------------------------------
__global__ void attn_out_res(const _Float16* __restrict__ rs, const _Float16* __restrict__ WoT,
                             const float* __restrict__ xin, float* __restrict__ xout)
{
    int lane = threadIdx.x & 31;
    int wave = threadIdx.x >> 5;      // 8 waves
    int mt = blockIdx.y;              // 0..127
    int nt = blockIdx.x * 8 + wave;   // 0..511
    int row0 = mt * 16, n0 = nt * 16;
    int m  = lane & 15;
    int kb = (lane >> 4) * 8;
    v8f acc = zero_acc();
#pragma unroll
    for (int k0 = 0; k0 < SS; k0 += 32) {
        v16h a = load_frag(rs  + (size_t)(row0 + m) * SS + k0 + kb);
        v16h b = load_frag(WoT + (size_t)(n0   + m) * SS + k0 + kb);
        acc = WMMA_F16(a, b, acc);
    }
    int ncol = n0 + m;
#pragma unroll
    for (int v = 0; v < 8; ++v) {
        int mm = v + 8 * (lane >> 4);
        size_t idx = (size_t)(row0 + mm) * DD + ncol;
        xout[idx] = xin[idx] + acc[v];
    }
}

// ------------------------------------------------------------------
// K6: freq conv as 3 shifted GEMMs (M2,64)x(64,64) + GELU, residual
// ------------------------------------------------------------------
__global__ void freq_conv_res(const _Float16* __restrict__ xn, const _Float16* __restrict__ fW,
                              const float* __restrict__ bias, float* __restrict__ xio)
{
    int lane = threadIdx.x & 31;
    int wave = threadIdx.x >> 5;      // 4 waves -> N tile
    int row0 = blockIdx.x * 16;
    int m  = lane & 15;
    int kb = (lane >> 4) * 8;
    int n0 = wave * 16;
    int rowA = row0 + m;
    int f = rowA & (FF - 1);
    v8f acc = zero_acc();
#pragma unroll
    for (int k = 0; k < 3; ++k) {
        int shift = k - 1;
        int fp = f + shift;
        bool valid = (fp >= 0) && (fp < FF);
        const _Float16* Arow = xn + (size_t)(rowA + shift) * HH;
        const _Float16* Brow = fW + (size_t)k * (HH * HH) + (size_t)(n0 + m) * HH;
#pragma unroll
        for (int k0 = 0; k0 < HH; k0 += 32) {
            v16h a = valid ? load_frag(Arow + k0 + kb) : zero_frag();
            v16h b = load_frag(Brow + k0 + kb);
            acc = WMMA_F16(a, b, acc);
        }
    }
    int ncol = n0 + m;
    float bn = bias[ncol];
#pragma unroll
    for (int v = 0; v < 8; ++v) {
        int mm = v + 8 * (lane >> 4);
        xio[(size_t)(row0 + mm) * HH + ncol] += gelu_tanh(acc[v] + bn);
    }
}

// ------------------------------------------------------------------
// K8: fused causal depthwise time-conv + LayerNorm -> delta buffer
// ------------------------------------------------------------------
__global__ void time_conv_ln(const float* __restrict__ xio, const float* __restrict__ st,
                             const float* __restrict__ lng, const float* __restrict__ lnb,
                             const float* __restrict__ tW, const float* __restrict__ tb,
                             const float* __restrict__ g2, const float* __restrict__ b2,
                             float* __restrict__ tdelta)
{
    __shared__ float rowbuf[DD];
    __shared__ float r1[256], r2[256];
    int tid = threadIdx.x;
    int bt = blockIdx.x;
    int b = bt >> 9, t = bt & (TT - 1);
    float ps = 0.f, pq = 0.f;
    for (int d = tid; d < DD; d += 256) {
        float g = lng[d], bb = lnb[d];
        float acc = tb[d];
#pragma unroll
        for (int k = 0; k < 3; ++k) {
            int tp = t - 2 + k;
            if (tp >= 0) {
                int rr = (b << 9) + tp;
                float xv = xio[(size_t)rr * DD + d];
                float xn = (xv - st[rr * 2]) * st[rr * 2 + 1] * g + bb;
                acc += xn * tW[d * 3 + k];
            }
        }
        rowbuf[d] = acc;
        ps += acc; pq += acc * acc;
    }
    r1[tid] = ps; r2[tid] = pq; __syncthreads();
    for (int off = 128; off > 0; off >>= 1) {
        if (tid < off) { r1[tid] += r1[tid + off]; r2[tid] += r2[tid + off]; }
        __syncthreads();
    }
    float mu = r1[0] * (1.f / DD);
    float rstd = rsqrtf(r2[0] * (1.f / DD) - mu * mu + 1e-5f);
    for (int d = tid; d < DD; d += 256)
        tdelta[(size_t)bt * DD + d] = (rowbuf[d] - mu) * rstd * g2[d] + b2[d];
}

// ------------------------------------------------------------------
// K9: FFN on H with LDS-staged mid tile; fuses stage-3 delta + final out
// ------------------------------------------------------------------
__global__ void ffn_res(const float* __restrict__ xio, const float* __restrict__ tdel,
                        const _Float16* __restrict__ W1T, const float* __restrict__ b1,
                        const _Float16* __restrict__ W2T, const float* __restrict__ b2,
                        float* __restrict__ xout)
{
    __shared__ _Float16 As[16 * HH];      // 2 KB
    __shared__ float    Asf[16 * HH];     // 4 KB
    __shared__ _Float16 Mid[16 * H4];     // 8 KB
    int tid = threadIdx.x;                // 128 threads = 4 waves
    int lane = tid & 31, wave = tid >> 5;
    size_t row0 = (size_t)blockIdx.x * 16;
    for (int i = tid; i < 16 * HH; i += 128) {
        float xv = xio[row0 * HH + i] + tdel[row0 * HH + i];   // x after stage 3
        Asf[i] = xv;
        As[i] = (_Float16)xv;
    }
    __syncthreads();
    int m = lane & 15, kb = (lane >> 4) * 8;
    // GEMM1: (16,64)x(64,256) + GELU -> Mid (f16)
#pragma unroll
    for (int sub = 0; sub < 4; ++sub) {
        int n0 = wave * 64 + sub * 16;
        v8f acc = zero_acc();
#pragma unroll
        for (int k0 = 0; k0 < HH; k0 += 32) {
            v16h a = load_frag(As  + m * HH + k0 + kb);
            v16h b = load_frag(W1T + (n0 + m) * HH + k0 + kb);
            acc = WMMA_F16(a, b, acc);
        }
        int ncol = n0 + m;
        float bb = b1[ncol];
#pragma unroll
        for (int v = 0; v < 8; ++v) {
            int mm = v + 8 * (lane >> 4);
            Mid[mm * H4 + ncol] = (_Float16)gelu_tanh(acc[v] + bb);
        }
    }
    __syncthreads();
    // GEMM2: (16,256)x(256,64); write final x + ffn
    {
        int n0 = wave * 16;
        v8f acc = zero_acc();
#pragma unroll
        for (int k0 = 0; k0 < H4; k0 += 32) {
            v16h a = load_frag(Mid + m * H4 + k0 + kb);
            v16h b = load_frag(W2T + (n0 + m) * H4 + k0 + kb);
            acc = WMMA_F16(a, b, acc);
        }
        int ncol = n0 + m;
        float bb = b2[ncol];
#pragma unroll
        for (int v = 0; v < 8; ++v) {
            int mm = v + 8 * (lane >> 4);
            xout[(row0 + mm) * HH + ncol] = Asf[mm * HH + ncol] + acc[v] + bb;
        }
    }
}

// ------------------------------------------------------------------
extern "C" void kernel_launch(void* const* d_in, const int* in_sizes, int n_in,
                              void* d_out, int out_size, void* d_ws, size_t ws_size,
                              hipStream_t stream) {
    const float* x         = (const float*)d_in[0];
    const float* ln_attn_g = (const float*)d_in[1];
    const float* ln_attn_b = (const float*)d_in[2];
    const float* ln_freq_g = (const float*)d_in[3];
    const float* ln_freq_b = (const float*)d_in[4];
    const float* ln_time_g = (const float*)d_in[5];
    const float* ln_time_b = (const float*)d_in[6];
    const float* Wk        = (const float*)d_in[7];
    const float* Wv        = (const float*)d_in[8];
    const float* Wr        = (const float*)d_in[9];
    const float* Wo        = (const float*)d_in[10];
    const float* w_decay   = (const float*)d_in[11];
    const float* freq_W    = (const float*)d_in[12];
    const float* freq_b    = (const float*)d_in[13];
    const float* time_W    = (const float*)d_in[14];
    const float* time_b    = (const float*)d_in[15];
    const float* time_ln_g = (const float*)d_in[16];
    const float* time_ln_b = (const float*)d_in[17];
    const float* ffn_W1    = (const float*)d_in[18];
    const float* ffn_b1    = (const float*)d_in[19];
    const float* ffn_W2    = (const float*)d_in[20];
    const float* ffn_b2    = (const float*)d_in[21];
    float* xcur = (float*)d_out;                  // running residual tensor

    // ---- workspace carve-out (256B aligned) ----
    char* w = (char*)d_ws;
    auto carve = [&](size_t bytes) { char* p = w; w += (bytes + 255) & ~(size_t)255; return p; };
    _Float16* xn_h  = (_Float16*)carve((size_t)M1 * DD * 2);
    float*    kbuf  = (float*)   carve((size_t)M1 * SS * 4);
    float*    vbuf  = (float*)   carve((size_t)M1 * SS * 4);
    float*    rbuf  = (float*)   carve((size_t)M1 * SS * 4);
    _Float16* rs_h  = (_Float16*)carve((size_t)M1 * SS * 2);
    float*    stats = (float*)   carve((size_t)M1 * 2 * 4);
    float*    tdel  = (float*)   carve((size_t)M1 * DD * 4);
    _Float16* WkT   = (_Float16*)carve((size_t)SS * DD * 2);
    _Float16* WvT   = (_Float16*)carve((size_t)SS * DD * 2);
    _Float16* WrT   = (_Float16*)carve((size_t)SS * DD * 2);
    _Float16* WoT   = (_Float16*)carve((size_t)DD * SS * 2);
    _Float16* fWh   = (_Float16*)carve((size_t)3 * HH * HH * 2);
    _Float16* W1T   = (_Float16*)carve((size_t)H4 * HH * 2);
    _Float16* W2T   = (_Float16*)carve((size_t)HH * H4 * 2);

    // K0: weights -> f16 transposed
    {
        int total = 4 * SS * DD + 3 * HH * HH + H4 * HH + HH * H4;
        prep_weights<<<(total + 255) / 256, 256, 0, stream>>>(
            Wk, Wv, Wr, Wo, freq_W, ffn_W1, ffn_W2,
            WkT, WvT, WrT, WoT, fWh, W1T, W2T);
    }
    // ---- stage 1: RWKV time mixing ----
    ln_to_f16<<<M1, 256, 0, stream>>>(x, ln_attn_g, ln_attn_b, xn_h);
    proj_kvr<<<M1 / 16, 384, 0, stream>>>(xn_h, WkT, WvT, WrT, kbuf, vbuf, rbuf);
    rwkv_scan<<<BZ, SS, 0, stream>>>(kbuf, vbuf, rbuf, w_decay, rs_h);
    attn_out_res<<<dim3(64, M1 / 16), 256, 0, stream>>>(rs_h, WoT, x, xcur);
    // ---- stage 2: freq conv ----
    ln_to_f16<<<M1, 256, 0, stream>>>(xcur, ln_freq_g, ln_freq_b, xn_h);
    freq_conv_res<<<M2 / 16, 128, 0, stream>>>(xn_h, fWh, freq_b, xcur);
    // ---- stage 3: causal depthwise time conv + LN -> delta ----
    row_stats<<<M1, 256, 0, stream>>>(xcur, stats);
    time_conv_ln<<<M1, 256, 0, stream>>>(xcur, stats, ln_time_g, ln_time_b,
                                         time_W, time_b, time_ln_g, time_ln_b, tdel);
    // ---- stage 4: FFN on H (fuses stage-3 residual add + final write) ----
    ffn_res<<<M2 / 16, 128, 0, stream>>>(xcur, tdel, W1T, ffn_b1, W2T, ffn_b2, xcur);
}
